// SelfAttention_41764261986688
// MI455X (gfx1250) — compile-verified, matrix operand order checked by reading, not run on hardware
//
#include <hip/hip_runtime.h>
#include <hip/hip_bf16.h>

typedef __bf16 bf16_t;
typedef __attribute__((ext_vector_type(16))) __bf16 v16bf;
typedef __attribute__((ext_vector_type(8)))  __bf16 v8bf;
typedef __attribute__((ext_vector_type(8)))  float  v8f;

#define LDP     72        // LDS row stride (elements) = 64 + 8 pad (bank-conflict-free, 16B aligned)
#define T_SEQ   2048
#define NH      16
#define HD      64
#define DMODEL  1024
#define NB      2
#define SCALE_Q 0.125f    // 64^-0.5

__device__ __forceinline__ v8f wmma_bf16(v16bf a, v16bf b, v8f c) {
  // D = A(16x32) * B(32x16) + C, f32 accumulate
  return __builtin_amdgcn_wmma_f32_16x16x32_bf16(false, a, false, b, (short)0, c, false, false);
}

// CDNA5 async copy: global (16B per lane) -> LDS, tracked on ASYNCcnt.
// Generic pointers to __shared__ carry the LDS offset in their low 32 bits
// (ISA aperture rule).
__device__ __forceinline__ void async_load_b128(const void* gsrc, void* ldst) {
  asm volatile("global_load_async_to_lds_b128 %0, %1, off"
               :: "v"((unsigned)(size_t)ldst),
                  "v"((unsigned long long)(size_t)gsrc)
               : "memory");
}
__device__ __forceinline__ void wait_async0() {
  asm volatile("s_wait_asynccnt 0x0" ::: "memory");
}

// Load a 16x32 bf16 A/B fragment from an LDS tile stored row-major with stride LDP.
// 16-bit A layout: lanes 0-15 -> row=lane, elems[0..7]=K0..7, [8..15]=K16..23;
// lanes 16-31 -> row=lane-16, K8..15 and K24..31. Two ds_load_b128 per fragment.
__device__ __forceinline__ v16bf lds_frag(const bf16_t* tile, int rbase, int kbase) {
  const int lane = threadIdx.x & 31;
  const int lr = lane & 15, hf = lane >> 4;
  const bf16_t* p = tile + (rbase + lr) * LDP + kbase + hf * 8;
  union { v16bf v; v8bf h[2]; } u;
  u.h[0] = *(const v8bf*)p;
  u.h[1] = *(const v8bf*)(p + 16);
  return u.v;
}

// ---------------------------------------------------------------------------
// Generic GEMM: C(MxN) = A(MxK) @ W(KxN) + bias.
// Block tile 128x128, K-tile 64, 256 threads (8 waves: 4(M) x 2(N)),
// each wave owns a 32x64 region = 2x4 WMMA tiles.
// QKV_STORE: scatter bf16 into per-head [b][h][t][d] layout, bias + Q-scale folded.
// ---------------------------------------------------------------------------
template <bool A_IS_BF16, bool QKV_STORE>
__global__ void __launch_bounds__(256) gemm_wmma(
    const void* __restrict__ Aptr, const float* __restrict__ W,
    const float* __restrict__ bias, float* __restrict__ Cf,
    bf16_t* __restrict__ qo, bf16_t* __restrict__ ko, bf16_t* __restrict__ vo,
    int M, int N, int K)
{
  __shared__ __align__(16) bf16_t sA[128 * LDP];   // [m][k]
  __shared__ __align__(16) bf16_t sB[128 * LDP];   // [n][k] (W transposed)
  const int tid    = threadIdx.x;
  const int waveId = tid >> 5;
  const int waveM  = waveId >> 1;   // 0..3
  const int waveN  = waveId & 1;    // 0..1
  const int rowBase = blockIdx.y * 128;
  const int nBase   = blockIdx.x * 128;

  const v8f vzero = {0.f, 0.f, 0.f, 0.f, 0.f, 0.f, 0.f, 0.f};
  v8f acc[2][4];
  #pragma unroll
  for (int i = 0; i < 2; ++i)
    #pragma unroll
    for (int j = 0; j < 4; ++j) acc[i][j] = vzero;

  #pragma unroll 1
  for (int k0 = 0; k0 < K; k0 += 64) {
    // ---- A tile: 128 x 64 -> bf16 LDS ----
    #pragma unroll
    for (int i = 0; i < 8; ++i) {
      int idx = tid + i * 256;          // 0..2047 : 16 float4 per row
      int row = idx >> 4;
      int c   = (idx & 15) << 2;
      if (A_IS_BF16) {
        const bf16_t* src = (const bf16_t*)Aptr + (size_t)(rowBase + row) * K + k0 + c;
        sA[row * LDP + c + 0] = src[0];
        sA[row * LDP + c + 1] = src[1];
        sA[row * LDP + c + 2] = src[2];
        sA[row * LDP + c + 3] = src[3];
      } else {
        const float* src = (const float*)Aptr + (size_t)(rowBase + row) * K + k0 + c;
        const float4 f = *(const float4*)src;
        sA[row * LDP + c + 0] = (bf16_t)f.x;
        sA[row * LDP + c + 1] = (bf16_t)f.y;
        sA[row * LDP + c + 2] = (bf16_t)f.z;
        sA[row * LDP + c + 3] = (bf16_t)f.w;
      }
      if (k0 + 64 < K)
        __builtin_prefetch((const float*)Aptr + (size_t)(rowBase + row) * K + k0 + 64 + c, 0, 1);
    }
    // ---- W tile: 64(k) x 128(n) fp32 -> bf16 LDS transposed [n][k] ----
    #pragma unroll
    for (int i = 0; i < 8; ++i) {
      int idx = tid + i * 256;          // 32 float4 per k-row
      int kr  = idx >> 5;
      int c   = (idx & 31) << 2;
      const float4 f = *(const float4*)(W + (size_t)(k0 + kr) * N + nBase + c);
      sB[(c + 0) * LDP + kr] = (bf16_t)f.x;
      sB[(c + 1) * LDP + kr] = (bf16_t)f.y;
      sB[(c + 2) * LDP + kr] = (bf16_t)f.z;
      sB[(c + 3) * LDP + kr] = (bf16_t)f.w;
      if (k0 + 64 < K)
        __builtin_prefetch(W + (size_t)(k0 + 64 + kr) * N + nBase + c, 0, 1);
    }
    __syncthreads();
    #pragma unroll
    for (int kk = 0; kk < 64; kk += 32) {
      v16bf af[2];
      #pragma unroll
      for (int tm = 0; tm < 2; ++tm)
        af[tm] = lds_frag(sA, waveM * 32 + tm * 16, kk);
      #pragma unroll
      for (int tn = 0; tn < 4; ++tn) {
        v16bf bfr = lds_frag(sB, waveN * 64 + tn * 16, kk);
        #pragma unroll
        for (int tm = 0; tm < 2; ++tm)
          acc[tm][tn] = wmma_bf16(af[tm], bfr, acc[tm][tn]);
      }
    }
    __syncthreads();
  }

  // ---- store: C/D layout: element(m,n): lane = n + 16*(m>=8), vgpr = m%8 ----
  const int lane = tid & 31, lr = lane & 15, hf = lane >> 4;
  #pragma unroll
  for (int tm = 0; tm < 2; ++tm) {
    #pragma unroll
    for (int tn = 0; tn < 4; ++tn) {
      const int gn = nBase + waveN * 64 + tn * 16 + lr;
      const float bv = bias[gn];
      const int gmBase = rowBase + waveM * 32 + tm * 16 + hf * 8;
      #pragma unroll
      for (int r = 0; r < 8; ++r) {
        float val = acc[tm][tn][r] + bv;
        const int grow = gmBase + r;
        if (QKV_STORE) {
          const int s = gn >> 10, rem = gn & 1023, h = rem >> 6, d = rem & 63;
          if (s == 0) val *= SCALE_Q;
          bf16_t* dst = (s == 0) ? qo : (s == 1) ? ko : vo;
          const int bb = grow >> 11, t = grow & (T_SEQ - 1);
          dst[((size_t)(bb * NH + h) * T_SEQ + t) * HD + d] = (bf16_t)val;
        } else {
          Cf[(size_t)grow * N + gn] = val;
        }
      }
    }
  }
}

// ---------------------------------------------------------------------------
// Flash attention per (b, h, 128 q-rows). Each of 8 waves owns 16 q-rows.
// Streams 64-wide K/V blocks; online softmax; writes O (B,T,D) bf16 + stats.
// Q/K tiles filled via CDNA5 async global->LDS DMA; next block prefetched.
// ---------------------------------------------------------------------------
__global__ void __launch_bounds__(256) flash_attn(
    const bf16_t* __restrict__ Q, const bf16_t* __restrict__ Km,
    const bf16_t* __restrict__ Vm, bf16_t* __restrict__ Oh,
    float* __restrict__ Ms, float* __restrict__ Ls)
{
  __shared__ __align__(16) bf16_t sQ [128 * LDP];      // [q][d]
  __shared__ __align__(16) bf16_t sK [64 * LDP];       // [kc][d]
  __shared__ __align__(16) bf16_t sVt[64 * LDP];       // [d][kc]
  __shared__ __align__(16) bf16_t sP [8 * 16 * LDP];   // per-wave P scratch [q][kc]
  const int tid = threadIdx.x;
  const int waveId = tid >> 5;
  const int lane = tid & 31, lr = lane & 15, hf = lane >> 4;
  const int q0 = blockIdx.x * 128;
  const int h  = blockIdx.y;
  const int b  = blockIdx.z;
  const size_t headBase = (size_t)(b * NH + h) * T_SEQ * HD;

  // Q tile resident for the whole kernel: async DMA to LDS
  #pragma unroll
  for (int i = 0; i < 4; ++i) {
    int idx = tid + i * 256;            // 128 rows x 8 chunks
    int row = idx >> 3;
    int c   = (idx & 7) << 3;
    async_load_b128(&Q[headBase + (size_t)(q0 + row) * HD + c], &sQ[row * LDP + c]);
  }
  wait_async0();
  __syncthreads();

  const v8f vzero = {0.f, 0.f, 0.f, 0.f, 0.f, 0.f, 0.f, 0.f};
  v8f o[4];
  #pragma unroll
  for (int i = 0; i < 4; ++i) o[i] = vzero;
  float mrun[8], lrun[8];
  #pragma unroll
  for (int r = 0; r < 8; ++r) { mrun[r] = -3.0e38f; lrun[r] = 0.f; }

  #pragma unroll 1
  for (int kb = 0; kb < T_SEQ / 64; ++kb) {
    const int kt0 = kb * 64;
    #pragma unroll
    for (int i = 0; i < 2; ++i) {
      int idx = tid + i * 256;          // 64 rows x 8 chunks
      int row = idx >> 3;
      int c   = (idx & 7) << 3;
      async_load_b128(&Km[headBase + (size_t)(kt0 + row) * HD + c], &sK[row * LDP + c]);
      v8bf vv = *(const v8bf*)&Vm[headBase + (size_t)(kt0 + row) * HD + c];
      #pragma unroll
      for (int j = 0; j < 8; ++j) sVt[(c + j) * LDP + row] = vv[j];
      if (kt0 + 64 < T_SEQ) {
        __builtin_prefetch(&Km[headBase + (size_t)(kt0 + 64 + row) * HD + c], 0, 1);
        __builtin_prefetch(&Vm[headBase + (size_t)(kt0 + 64 + row) * HD + c], 0, 1);
      }
    }
    wait_async0();
    __syncthreads();

    // S = Q @ K^T : 16(q) x 64(kc), K-dim = 64 -> 2 wmma steps x 4 n-tiles
    v8f s[4];
    #pragma unroll
    for (int i = 0; i < 4; ++i) s[i] = vzero;
    #pragma unroll
    for (int kk = 0; kk < 64; kk += 32) {
      v16bf aq = lds_frag(sQ, waveId * 16, kk);
      #pragma unroll
      for (int tn = 0; tn < 4; ++tn) {
        v16bf bk = lds_frag(sK, tn * 16, kk);
        s[tn] = wmma_bf16(aq, bk, s[tn]);
      }
    }

    // Online softmax: each q-row lives in one 16-lane half, one vgpr index r.
    #pragma unroll
    for (int r = 0; r < 8; ++r) {
      float m = fmaxf(fmaxf(s[0][r], s[1][r]), fmaxf(s[2][r], s[3][r]));
      #pragma unroll
      for (int d = 1; d < 16; d <<= 1) m = fmaxf(m, __shfl_xor(m, d, 32));
      const float mnew = fmaxf(mrun[r], m);
      const float corr = __expf(mrun[r] - mnew);
      mrun[r] = mnew;
      float sum = 0.f;
      #pragma unroll
      for (int tn = 0; tn < 4; ++tn) {
        float p = __expf(s[tn][r] - mnew);
        s[tn][r] = p;
        sum += p;
      }
      #pragma unroll
      for (int d = 1; d < 16; d <<= 1) sum += __shfl_xor(sum, d, 32);
      lrun[r] = lrun[r] * corr + sum;
      #pragma unroll
      for (int tn = 0; tn < 4; ++tn) o[tn][r] *= corr;
    }

    // Re-shape P (D-layout) into A-fragment layout via per-wave LDS scratch.
    bf16_t* sPw = &sP[waveId * 16 * LDP];
    #pragma unroll
    for (int tn = 0; tn < 4; ++tn)
      #pragma unroll
      for (int r = 0; r < 8; ++r)
        sPw[(r + 8 * hf) * LDP + tn * 16 + lr] = (bf16_t)s[tn][r];
    asm volatile("s_wait_dscnt 0" ::: "memory");   // wave-private RAW through LDS

    // O += P @ V : A = P(16x64), B = Vt[d][kc]
    #pragma unroll
    for (int kk = 0; kk < 64; kk += 32) {
      v16bf ap = lds_frag(sPw, 0, kk);
      #pragma unroll
      for (int tn = 0; tn < 4; ++tn) {
        v16bf bv = lds_frag(sVt, tn * 16, kk);
        o[tn] = wmma_bf16(ap, bv, o[tn]);
      }
    }
    __syncthreads();
  }

  // Finalize: O /= l ; store in (B,T,D) bf16; store softmax stats.
  #pragma unroll
  for (int r = 0; r < 8; ++r) {
    const float inv = 1.0f / lrun[r];
    const int t = q0 + waveId * 16 + r + 8 * hf;
    #pragma unroll
    for (int tn = 0; tn < 4; ++tn) {
      const int col = h * HD + tn * 16 + lr;
      Oh[(size_t)(b * T_SEQ + t) * DMODEL + col] = (bf16_t)(o[tn][r] * inv);
    }
    if (lr == 0) {
      const size_t si = (size_t)(b * NH + h) * T_SEQ + t;
      Ms[si] = mrun[r];
      Ls[si] = lrun[r];
    }
  }
}

// ---------------------------------------------------------------------------
// Head-averaged attention probabilities: recompute S per head with WMMA,
// apply exact softmax using stats from flash_attn, accumulate mean over heads.
// Block = (b, 128 q-rows, 64 k-cols). Q/K tiles via async global->LDS DMA.
// ---------------------------------------------------------------------------
__global__ void __launch_bounds__(256) attn_weights(
    const bf16_t* __restrict__ Q, const bf16_t* __restrict__ Km,
    const float* __restrict__ Ms, const float* __restrict__ Ls,
    float* __restrict__ AW)
{
  __shared__ __align__(16) bf16_t sQ[128 * LDP];
  __shared__ __align__(16) bf16_t sK[64 * LDP];
  __shared__ float sM[128];
  __shared__ float sIL[128];
  const int tid = threadIdx.x;
  const int waveId = tid >> 5;
  const int lane = tid & 31, lr = lane & 15, hf = lane >> 4;
  const int k0 = blockIdx.x * 64;
  const int q0 = blockIdx.y * 128;
  const int b  = blockIdx.z;

  const v8f vzero = {0.f, 0.f, 0.f, 0.f, 0.f, 0.f, 0.f, 0.f};
  v8f aw[4];
  #pragma unroll
  for (int i = 0; i < 4; ++i) aw[i] = vzero;

  #pragma unroll 1
  for (int h = 0; h < NH; ++h) {
    const size_t headBase = (size_t)(b * NH + h) * T_SEQ * HD;
    #pragma unroll
    for (int i = 0; i < 4; ++i) {
      int idx = tid + i * 256;
      int row = idx >> 3;
      int c   = (idx & 7) << 3;
      async_load_b128(&Q[headBase + (size_t)(q0 + row) * HD + c], &sQ[row * LDP + c]);
    }
    #pragma unroll
    for (int i = 0; i < 2; ++i) {
      int idx = tid + i * 256;
      int row = idx >> 3;
      int c   = (idx & 7) << 3;
      async_load_b128(&Km[headBase + (size_t)(k0 + row) * HD + c], &sK[row * LDP + c]);
    }
    if (tid < 128) {
      const size_t si = (size_t)(b * NH + h) * T_SEQ + q0 + tid;
      sM[tid]  = Ms[si];
      sIL[tid] = 1.0f / Ls[si];
    }
    wait_async0();
    __syncthreads();

    v8f s[4];
    #pragma unroll
    for (int i = 0; i < 4; ++i) s[i] = vzero;
    #pragma unroll
    for (int kk = 0; kk < 64; kk += 32) {
      v16bf aq = lds_frag(sQ, waveId * 16, kk);
      #pragma unroll
      for (int tn = 0; tn < 4; ++tn) {
        v16bf bk = lds_frag(sK, tn * 16, kk);
        s[tn] = wmma_bf16(aq, bk, s[tn]);
      }
    }
    #pragma unroll
    for (int r = 0; r < 8; ++r) {
      const int m = waveId * 16 + r + 8 * hf;
      const float mr = sM[m], il = sIL[m];
      #pragma unroll
      for (int tn = 0; tn < 4; ++tn)
        aw[tn][r] += __expf(s[tn][r] - mr) * il;
    }
    __syncthreads();
  }

  #pragma unroll
  for (int r = 0; r < 8; ++r) {
    const int qrow = q0 + waveId * 16 + r + 8 * hf;
    #pragma unroll
    for (int tn = 0; tn < 4; ++tn) {
      const int kc = k0 + tn * 16 + lr;
      AW[((size_t)b * T_SEQ + qrow) * T_SEQ + kc] = aw[tn][r] * (1.0f / NH);
    }
  }
}

// ---------------------------------------------------------------------------
extern "C" void kernel_launch(void* const* d_in, const int* in_sizes, int n_in,
                              void* d_out, int out_size, void* d_ws, size_t ws_size,
                              hipStream_t stream) {
  const float* x      = (const float*)d_in[0];
  const float* w_qkv  = (const float*)d_in[1];
  const float* b_qkv  = (const float*)d_in[2];
  const float* w_proj = (const float*)d_in[3];
  const float* b_proj = (const float*)d_in[4];

  float* out = (float*)d_out;                              // (B,T,D) fp32
  float* aw  = out + (size_t)NB * T_SEQ * DMODEL;          // (B,T,T) fp32

  // workspace layout (bf16 unless noted): qs | ks | vs | oh | Ms(f32) | Ls(f32)
  const size_t nQKV = (size_t)NB * NH * T_SEQ * HD;        // 4M elems each
  bf16_t* qs = (bf16_t*)d_ws;
  bf16_t* ks = qs + nQKV;
  bf16_t* vs = ks + nQKV;
  bf16_t* oh = vs + nQKV;                                  // (B,T,D) bf16
  float*  Ms = (float*)(oh + nQKV);
  float*  Ls = Ms + (size_t)NB * NH * T_SEQ;

  // 1) QKV projection (scale folded into Q, bias folded in, per-head scatter)
  gemm_wmma<false, true><<<dim3(3 * DMODEL / 128, NB * T_SEQ / 128), 256, 0, stream>>>(
      x, w_qkv, b_qkv, nullptr, qs, ks, vs, NB * T_SEQ, 3 * DMODEL, DMODEL);

  // 2) Flash attention -> oh (B,T,D) bf16 + row stats
  flash_attn<<<dim3(T_SEQ / 128, NH, NB), 256, 0, stream>>>(qs, ks, vs, oh, Ms, Ls);

  // 3) Head-averaged attention probabilities
  attn_weights<<<dim3(T_SEQ / 64, T_SEQ / 128, NB), 256, 0, stream>>>(qs, ks, Ms, Ls, aw);

  // 4) Output projection -> d_out fp32
  gemm_wmma<true, false><<<dim3(DMODEL / 128, NB * T_SEQ / 128), 256, 0, stream>>>(
      oh, w_proj, b_proj, out, nullptr, nullptr, nullptr, NB * T_SEQ, DMODEL, DMODEL);
}
